// CausalSelfAttention_43319040147880
// MI455X (gfx1250) — matrix operand order repaired and checked by reading, not compiled, not run on hardware
//
#include <hip/hip_runtime.h>
#include <math.h>

#define S_LEN 2048
#define C_DIM 1024
#define HEADS 16
#define HDIM  64
#define BATCH 2
#define M_ROWS (BATCH * S_LEN)   // 4096
#define N3C    (3 * C_DIM)       // 3072

typedef __attribute__((ext_vector_type(16))) __bf16 bf16x16;
typedef __attribute__((ext_vector_type(8)))  float  f32x8;

union BF8 { uint4 u; __bf16 h[8]; };

__device__ __forceinline__ unsigned short f2bf(float f) {
  union { float f; unsigned u; } v; v.f = f;
  unsigned r = v.u + 0x7FFFu + ((v.u >> 16) & 1u);  // round-to-nearest-even
  return (unsigned short)(r >> 16);
}

// Build a 16-element bf16 A/B fragment from two 16-byte chunks.
__device__ __forceinline__ bf16x16 frag_from2(const unsigned short* p0,
                                              const unsigned short* p1) {
  BF8 a, b;
  a.u = *(const uint4*)p0;
  b.u = *(const uint4*)p1;
  bf16x16 r;
#pragma unroll
  for (int i = 0; i < 8; ++i) { r[i] = a.h[i]; r[i + 8] = b.h[i]; }
  return r;
}

__device__ __forceinline__ f32x8 wmma_bf16(bf16x16 a, bf16x16 b, f32x8 c) {
  return __builtin_amdgcn_wmma_f32_16x16x32_bf16(false, a, false, b,
                                                 (short)0, c, false, false);
}

// ---------------- conversion kernels ----------------
__global__ void cvt_bf16(const float* __restrict__ in,
                         unsigned short* __restrict__ out, int n) {
  int i = blockIdx.x * blockDim.x + threadIdx.x;
  if (i < n) out[i] = f2bf(in[i]);
}

// in: f32 [K][N] row-major  ->  out: bf16 [N][K] (transposed)
__global__ void cvt_transpose(const float* __restrict__ in,
                              unsigned short* __restrict__ out, int K, int N) {
  int i = blockIdx.x * blockDim.x + threadIdx.x;
  if (i < K * N) {
    int k = i / N, n = i % N;
    out[(size_t)n * K + k] = f2bf(in[i]);
  }
}

// ---------------- GEMM1: qkv = x @ w_attn + b_attn, routed to Q/K/Vt ----------------
// Wave tile: 32(M) x 64(N), 8 WMMAs / k-step. Block: 8 waves = 64(M) x 256(N).
__global__ void __launch_bounds__(256)
gemm_qkv(const unsigned short* __restrict__ xb,    // bf16 [4096][1024]
         const unsigned short* __restrict__ wat,   // bf16 [3072][1024] (W^T)
         const float* __restrict__ bias,           // [3072]
         unsigned short* __restrict__ Q,           // bf16 [B,H,S,D]
         unsigned short* __restrict__ Kt,          // bf16 [B,H,S,D]
         unsigned short* __restrict__ Vt) {        // bf16 [B,H,D,S]
  const int lane  = threadIdx.x & 31;
  const int wave  = threadIdx.x >> 5;
  const int nBase = blockIdx.x * 256 + (wave & 3) * 64;
  const int mBase = blockIdx.y * 64  + (wave >> 2) * 32;
  const int ln    = lane & 15;
  const int aoff  = (lane < 16) ? 0 : 8;    // A-frag K-half
  const int boff  = (lane < 16) ? 0 : 8;    // B-frag K-half

  f32x8 acc[2][4];
#pragma unroll
  for (int i = 0; i < 2; ++i)
#pragma unroll
    for (int j = 0; j < 4; ++j)
      acc[i][j] = (f32x8){0.f,0.f,0.f,0.f,0.f,0.f,0.f,0.f};

  const unsigned short* ap0 = xb  + (size_t)(mBase + ln)      * C_DIM + aoff;
  const unsigned short* ap1 = xb  + (size_t)(mBase + 16 + ln) * C_DIM + aoff;
  const unsigned short* bp  = wat + (size_t)(nBase + ln)      * C_DIM + boff;

  for (int k0 = 0; k0 < C_DIM; k0 += 32) {
    bf16x16 a0 = frag_from2(ap0 + k0, ap0 + k0 + 16);
    bf16x16 a1 = frag_from2(ap1 + k0, ap1 + k0 + 16);
#pragma unroll
    for (int j = 0; j < 4; ++j) {
      const unsigned short* bj = bp + (size_t)(j * 16) * C_DIM + k0;
      bf16x16 b = frag_from2(bj, bj + 16);
      acc[0][j] = wmma_bf16(a0, b, acc[0][j]);
      acc[1][j] = wmma_bf16(a1, b, acc[1][j]);
    }
  }

#pragma unroll
  for (int j = 0; j < 4; ++j) {
    const int   col  = nBase + j * 16 + ln;
    const float bval = bias[col];
    const int   seg  = col >> 10;   // 0=Q, 1=K, 2=V
    const int   c    = col & 1023;
    const int   h    = c >> 6;
    const int   d    = c & 63;
#pragma unroll
    for (int i = 0; i < 2; ++i) {
      const int rbase = mBase + i * 16 + ((lane < 16) ? 0 : 8);
#pragma unroll
      for (int r = 0; r < 8; ++r) {
        const int row = rbase + r;
        const int bi  = row >> 11;           // batch
        const int s   = row & (S_LEN - 1);
        const unsigned short val = f2bf(acc[i][j][r] + bval);
        const size_t head = (size_t)(bi * HEADS + h);
        if (seg == 0)      Q [(head * S_LEN + s) * HDIM + d] = val;
        else if (seg == 1) Kt[(head * S_LEN + s) * HDIM + d] = val;
        else               Vt[(head * HDIM + d) * S_LEN + s] = val;
      }
    }
  }
}

// ---------------- flash attention: one wave per 16-query tile ----------------
__global__ void __launch_bounds__(32)
attn(const unsigned short* __restrict__ Q,
     const unsigned short* __restrict__ K,
     const unsigned short* __restrict__ Vt,
     unsigned short* __restrict__ Y) {            // bf16 [B,S,C]
  __shared__ unsigned short pl[16 * 32];          // P tile staging (C-layout -> A-layout)
  const int lane = threadIdx.x;
  const int qt = blockIdx.x & 127;                // S/16 tiles
  const int bh = blockIdx.x >> 7;                 // 0..31
  const int q0 = qt * 16;
  const unsigned short* Qb = Q  + (size_t)bh * S_LEN * HDIM;
  const unsigned short* Kb = K  + (size_t)bh * S_LEN * HDIM;
  const unsigned short* Vb = Vt + (size_t)bh * HDIM * S_LEN;
  const int ln   = lane & 15;
  const int half = (lane < 16) ? 0 : 8;           // A-frag K-half / C-frag row-half
  const int boff = (lane < 16) ? 0 : 8;           // B-frag K-half

  // Q A-fragments (D = 64 -> two 16x32 fragments)
  const unsigned short* qp = Qb + (size_t)(q0 + ln) * HDIM + half;
  bf16x16 qa0 = frag_from2(qp, qp + 16);
  bf16x16 qa1 = frag_from2(qp + 32, qp + 48);

  f32x8 o0 = {0.f,0.f,0.f,0.f,0.f,0.f,0.f,0.f};
  f32x8 o1 = o0, o2 = o0, o3 = o0;
  float m[8], l[8];
#pragma unroll
  for (int r = 0; r < 8; ++r) { m[r] = -1e30f; l[r] = 0.f; }
  const int rbase = q0 + half;

  for (int kc = 0; kc < q0 + 16; kc += 32) {
    // ---- scores for keys kc..kc+31 : 4 WMMAs ----
    f32x8 s0 = {0.f,0.f,0.f,0.f,0.f,0.f,0.f,0.f};
    f32x8 s1 = s0;
    {
      const unsigned short* kp = Kb + (size_t)(kc + ln) * HDIM + boff;
      bf16x16 kb0 = frag_from2(kp,      kp + 16);       // keys kc..kc+15, D 0..31
      bf16x16 kb1 = frag_from2(kp + 32, kp + 48);       // keys kc..kc+15, D 32..63
      s0 = wmma_bf16(qa0, kb0, s0);
      s0 = wmma_bf16(qa1, kb1, s0);
      const unsigned short* kq = kp + 16 * HDIM;        // keys kc+16..kc+31
      bf16x16 kb2 = frag_from2(kq,      kq + 16);
      bf16x16 kb3 = frag_from2(kq + 32, kq + 48);
      s1 = wmma_bf16(qa0, kb2, s1);
      s1 = wmma_bf16(qa1, kb3, s1);
    }
    // ---- online softmax (f32), causal mask ----
    const int col0 = kc + ln, col1 = col0 + 16;
#pragma unroll
    for (int r = 0; r < 8; ++r) {
      const int qrow = rbase + r;
      float v0 = s0[r] * 0.125f;
      float v1 = s1[r] * 0.125f;
      if (col0 > qrow) v0 = -1e30f;
      if (col1 > qrow) v1 = -1e30f;
      float mx = fmaxf(v0, v1);
      mx = fmaxf(mx, __shfl_xor(mx, 1, 32));
      mx = fmaxf(mx, __shfl_xor(mx, 2, 32));
      mx = fmaxf(mx, __shfl_xor(mx, 4, 32));
      mx = fmaxf(mx, __shfl_xor(mx, 8, 32));
      const float mn    = fmaxf(m[r], mx);
      const float alpha = __expf(m[r] - mn);
      m[r] = mn;
      const float p0 = __expf(v0 - mn);
      const float p1 = __expf(v1 - mn);
      float sum = p0 + p1;
      sum += __shfl_xor(sum, 1, 32);
      sum += __shfl_xor(sum, 2, 32);
      sum += __shfl_xor(sum, 4, 32);
      sum += __shfl_xor(sum, 8, 32);
      l[r] = l[r] * alpha + sum;
      o0[r] *= alpha; o1[r] *= alpha; o2[r] *= alpha; o3[r] *= alpha;
      pl[(r + half) * 32 + ln]      = f2bf(p0);
      pl[(r + half) * 32 + 16 + ln] = f2bf(p1);
    }
    asm volatile("s_wait_dscnt 0x0" ::: "memory");   // cross-lane LDS visibility
    // ---- reload P in A-layout: lane ln holds row ln ----
    BF8 x0, x1;
    x0.u = *(const uint4*)(pl + ln * 32 + half);
    x1.u = *(const uint4*)(pl + ln * 32 + half + 16);
    bf16x16 pa;
#pragma unroll
    for (int i = 0; i < 8; ++i) { pa[i] = x0.h[i]; pa[i + 8] = x1.h[i]; }
    // ---- O += P(16x32) . V(32x64) : 4 WMMAs ----
    const unsigned short* vp = Vb + (size_t)ln * S_LEN + kc + boff;
    o0 = wmma_bf16(pa, frag_from2(vp,              vp + 16),              o0);
    o1 = wmma_bf16(pa, frag_from2(vp + 16 * S_LEN, vp + 16 * S_LEN + 16), o1);
    o2 = wmma_bf16(pa, frag_from2(vp + 32 * S_LEN, vp + 32 * S_LEN + 16), o2);
    o3 = wmma_bf16(pa, frag_from2(vp + 48 * S_LEN, vp + 48 * S_LEN + 16), o3);
  }
  // ---- epilogue: normalize and store y (bf16) ----
  const int bi = bh >> 4, h = bh & 15;
#pragma unroll
  for (int r = 0; r < 8; ++r) {
    const float inv = 1.0f / l[r];
    const int s = rbase + r;
    unsigned short* yp = Y + (size_t)(bi * S_LEN + s) * C_DIM + h * HDIM;
    yp[ln]      = f2bf(o0[r] * inv);
    yp[16 + ln] = f2bf(o1[r] * inv);
    yp[32 + ln] = f2bf(o2[r] * inv);
    yp[48 + ln] = f2bf(o3[r] * inv);
  }
}

// ---------------- GEMM2: out = y @ w_proj + b_proj (f32 out) ----------------
// Same 32x64 wave tile / 64x256 block tile as GEMM1.
__global__ void __launch_bounds__(256)
gemm_proj(const unsigned short* __restrict__ yb,   // bf16 [4096][1024]
          const unsigned short* __restrict__ wpt,  // bf16 [1024][1024] (W^T)
          const float* __restrict__ bias,          // [1024]
          float* __restrict__ out) {               // f32 [4096][1024]
  const int lane  = threadIdx.x & 31;
  const int wave  = threadIdx.x >> 5;
  const int nBase = blockIdx.x * 256 + (wave & 3) * 64;
  const int mBase = blockIdx.y * 64  + (wave >> 2) * 32;
  const int ln    = lane & 15;
  const int aoff  = (lane < 16) ? 0 : 8;
  const int boff  = (lane < 16) ? 0 : 8;

  f32x8 acc[2][4];
#pragma unroll
  for (int i = 0; i < 2; ++i)
#pragma unroll
    for (int j = 0; j < 4; ++j)
      acc[i][j] = (f32x8){0.f,0.f,0.f,0.f,0.f,0.f,0.f,0.f};

  const unsigned short* ap0 = yb  + (size_t)(mBase + ln)      * C_DIM + aoff;
  const unsigned short* ap1 = yb  + (size_t)(mBase + 16 + ln) * C_DIM + aoff;
  const unsigned short* bp  = wpt + (size_t)(nBase + ln)      * C_DIM + boff;

  for (int k0 = 0; k0 < C_DIM; k0 += 32) {
    bf16x16 a0 = frag_from2(ap0 + k0, ap0 + k0 + 16);
    bf16x16 a1 = frag_from2(ap1 + k0, ap1 + k0 + 16);
#pragma unroll
    for (int j = 0; j < 4; ++j) {
      const unsigned short* bj = bp + (size_t)(j * 16) * C_DIM + k0;
      bf16x16 b = frag_from2(bj, bj + 16);
      acc[0][j] = wmma_bf16(a0, b, acc[0][j]);
      acc[1][j] = wmma_bf16(a1, b, acc[1][j]);
    }
  }

#pragma unroll
  for (int j = 0; j < 4; ++j) {
    const int   col  = nBase + j * 16 + ln;
    const float bval = bias[col];
#pragma unroll
    for (int i = 0; i < 2; ++i) {
      const int rbase = mBase + i * 16 + ((lane < 16) ? 0 : 8);
#pragma unroll
      for (int r = 0; r < 8; ++r)
        out[(size_t)(rbase + r) * C_DIM + col] = acc[i][j][r] + bval;
    }
  }
}

extern "C" void kernel_launch(void* const* d_in, const int* in_sizes, int n_in,
                              void* d_out, int out_size, void* d_ws, size_t ws_size,
                              hipStream_t stream) {
  const float* x      = (const float*)d_in[0];
  const float* w_attn = (const float*)d_in[1];
  const float* b_attn = (const float*)d_in[2];
  const float* w_proj = (const float*)d_in[3];
  const float* b_proj = (const float*)d_in[4];
  float* out = (float*)d_out;

  unsigned short* ws  = (unsigned short*)d_ws;
  unsigned short* xb  = ws;                                    // 4096*1024
  unsigned short* wat = xb  + (size_t)M_ROWS * C_DIM;          // 3072*1024
  unsigned short* wpt = wat + (size_t)N3C    * C_DIM;          // 1024*1024
  unsigned short* Q   = wpt + (size_t)C_DIM  * C_DIM;          // B*H*S*D
  unsigned short* Kw  = Q   + (size_t)M_ROWS * C_DIM;
  unsigned short* Vt  = Kw  + (size_t)M_ROWS * C_DIM;
  unsigned short* Y   = xb;  // x (bf16) is dead after gemm_qkv -> reuse for y

  cvt_bf16<<<(M_ROWS * C_DIM) / 256, 256, 0, stream>>>(x, xb, M_ROWS * C_DIM);
  cvt_transpose<<<(C_DIM * N3C + 255) / 256, 256, 0, stream>>>(w_attn, wat, C_DIM, N3C);
  cvt_transpose<<<(C_DIM * C_DIM + 255) / 256, 256, 0, stream>>>(w_proj, wpt, C_DIM, C_DIM);

  gemm_qkv<<<dim3(N3C / 256, M_ROWS / 64), 256, 0, stream>>>(xb, wat, b_attn, Q, Kw, Vt);
  attn<<<dim3(BATCH * HEADS * (S_LEN / 16)), 32, 0, stream>>>(Q, Kw, Vt, Y);
  gemm_proj<<<dim3(C_DIM / 256, M_ROWS / 64), 256, 0, stream>>>(Y, wpt, b_proj, out);
}
